// KNNSoftClassifier_80298708566723
// MI455X (gfx1250) — matrix operand order repaired and checked by reading, not compiled, not run on hardware
//
#include <hip/hip_runtime.h>
#include <hip/hip_bf16.h>
#include <stdint.h>

// ---------------- problem constants ----------------
#define NQ     2048
#define NF     100000
#define DIM    512
#define NFPAD  100096          // padded to multiple of 128
#define CCLS   1000
#define KTOP   32
#define TAU_F  0.2f
#define CHUNK  128
#define NCHUNK (NFPAD / CHUNK) // 782
#define NSLICE 4
#define CH_PER_SLICE 196       // ceil(782/4)
#define QTILE  16

typedef __attribute__((ext_vector_type(16))) __bf16 v16bf;
typedef __attribute__((ext_vector_type(8)))  float  v8f;

struct U128 { unsigned int x, y, z, w; };
struct U32x8 { U128 lo, hi; };   // 32 bytes == v16bf

// float -> bf16 round-to-nearest-even
__device__ __forceinline__ unsigned short f2bf(float f) {
    unsigned int u = __float_as_uint(f);
    unsigned int r = u + 0x7FFFu + ((u >> 16) & 1u);
    return (unsigned short)(r >> 16);
}

// ---------------- kernel 1/2: row L2-normalize + bf16 convert ----------------
// one wave per row; 8 rows per 256-thread block
__global__ __launch_bounds__(256)
void nrm_bf16_kernel(const float* __restrict__ src, unsigned short* __restrict__ dst,
                     int nrows, int nrows_pad) {
    const int wave = threadIdx.x >> 5;
    const int lane = threadIdx.x & 31;
    const int row  = blockIdx.x * 8 + wave;
    if (row >= nrows_pad) return;
    unsigned short* q = dst + (size_t)row * DIM;
    if (row >= nrows) {                   // zero the pad rows (sim == 0, never inserted)
        for (int j = 0; j < 16; ++j) q[lane + 32 * j] = 0;
        return;
    }
    const float* p = src + (size_t)row * DIM;
    float x[16]; float ss = 0.f;
    #pragma unroll
    for (int j = 0; j < 16; ++j) { x[j] = p[lane + 32 * j]; ss += x[j] * x[j]; }
    #pragma unroll
    for (int off = 16; off > 0; off >>= 1) ss += __shfl_xor(ss, off, 32);
    const float scale = 1.0f / fmaxf(sqrtf(ss), 1e-12f);   // matches x / max(||x||, eps)
    #pragma unroll
    for (int j = 0; j < 16; ++j) q[lane + 32 * j] = f2bf(x[j] * scale);
}

// ---------------- kernel 3: WMMA cosine GEMM + per-slice exact top-32 ----------------
// grid = (NQ/QTILE, NSLICE); block = 256 (8 waves)
__global__ __launch_bounds__(256)
void sim_topk_kernel(const unsigned short* __restrict__ qn,
                     const unsigned short* __restrict__ fn,
                     float* __restrict__ topV, int* __restrict__ topI) {
    __shared__ union __align__(16) {
        struct { unsigned short A[QTILE][DIM]; float S[QTILE][CHUNK]; } p1; // 24.0 KB
        struct { float V[8][512]; int I[8][512]; } p2;                      // 32.0 KB
    } sm;

    const int tid  = threadIdx.x;
    const int wave = tid >> 5, lane = tid & 31;
    const int half = lane >> 4, l16 = lane & 15;
    const int qtile = blockIdx.x;
    const int slice = blockIdx.y;

    // stage A tile (16 x 512 bf16 = 16 KB) into LDS
    {
        const U128* s = (const U128*)(qn + (size_t)qtile * QTILE * DIM);
        U128* d = (U128*)(&sm.p1.A[0][0]);
        for (int i = tid; i < (QTILE * DIM * 2) / 16; i += 256) d[i] = s[i];
    }
    __syncthreads();

    // per-thread running top-K (registers/scratch)
    float tv[KTOP]; int ti[KTOP];
    #pragma unroll
    for (int i = 0; i < KTOP; ++i) { tv[i] = -1e30f; ti[i] = 0; }
    float tmin = -1e30f;

    const int c0 = slice * CH_PER_SLICE;
    const int c1 = min(NCHUNK, c0 + CH_PER_SLICE);
    const int qrow = tid >> 4;   // 0..15 : query row for top-k phase
    const int ssub = tid & 15;   // 0..15 : 8 columns each

    for (int ch = c0; ch < c1; ++ch) {
        // ---- 16x16 tile of S via 16 chained bf16 WMMAs (K = 512) ----
        v8f acc = {};
        const int col = ch * CHUNK + wave * 16 + l16;          // this lane's B column
        const unsigned short* bbase = fn + (size_t)col * DIM + half * 8;
        #pragma unroll
        for (int k0 = 0; k0 < DIM / 32; ++k0) {
            // A fragment: lane half 0 -> K {k0*32+0..7, +16..23}; half 1 -> {+8..15, +24..31}
            const unsigned short* ap = &sm.p1.A[l16][k0 * 32 + half * 8];
            U32x8 at; at.lo = *(const U128*)ap; at.hi = *(const U128*)(ap + 16);
            // B fragment: lane half 0 -> K {0..7, 16..23}; half 1 -> {8..15, 24..31}
            const unsigned short* bp = bbase + k0 * 32;
            U32x8 bt; bt.lo = *(const U128*)bp; bt.hi = *(const U128*)(bp + 16);
            v16bf av = __builtin_bit_cast(v16bf, at);
            v16bf bv = __builtin_bit_cast(v16bf, bt);
            acc = __builtin_amdgcn_wmma_f32_16x16x32_bf16(false, av, false, bv,
                                                          (short)0, acc, false, false);
        }
        // D layout: VGPR r, lane -> row r + 8*half, col l16
        #pragma unroll
        for (int r = 0; r < 8; ++r)
            sm.p1.S[r + 8 * half][wave * 16 + l16] = acc[r];
        __syncthreads();

        // ---- top-k update: 16 threads per query, 8 cols each ----
        #pragma unroll
        for (int j = 0; j < 8; ++j) {
            const int nl = ssub * 8 + j;
            const float v = sm.p1.S[qrow][nl];
            const int gidx = ch * CHUNK + nl;
            if (gidx < NF && v > tmin) {
                int p = KTOP - 1;
                while (p > 0 && tv[p - 1] < v) { tv[p] = tv[p - 1]; ti[p] = ti[p - 1]; --p; }
                tv[p] = v; ti[p] = gidx;
                tmin = tv[KTOP - 1];
            }
        }
        __syncthreads();
    }

    // ---- merge 16 per-thread lists -> per-(query,slice) top-32, two rounds of 8 queries ----
    for (int r = 0; r < 2; ++r) {
        if ((qrow >> 3) == r) {
            const int lq = qrow & 7;
            #pragma unroll
            for (int i = 0; i < KTOP; ++i) {
                sm.p2.V[lq][ssub * KTOP + i] = tv[i];
                sm.p2.I[lq][ssub * KTOP + i] = ti[i];
            }
        }
        __syncthreads();
        {   // wave w extracts top-32 of 512 candidates for local query r*8+w
            const int gq = qtile * QTILE + r * 8 + wave;
            float* outV = topV + (size_t)gq * (NSLICE * KTOP) + slice * KTOP;
            int*   outIp = topI + (size_t)gq * (NSLICE * KTOP) + slice * KTOP;
            for (int iter = 0; iter < KTOP; ++iter) {
                float bv = -1e30f; int bi = 0x7fffffff; int be = 0;
                #pragma unroll
                for (int j = 0; j < 16; ++j) {
                    const int e = lane + 32 * j;
                    const float v = sm.p2.V[wave][e];
                    const int  ix = sm.p2.I[wave][e];
                    if (v > bv || (v == bv && ix < bi)) { bv = v; bi = ix; be = e; }
                }
                #pragma unroll
                for (int off = 16; off > 0; off >>= 1) {
                    const float ov = __shfl_xor(bv, off, 32);
                    const int   oi = __shfl_xor(bi, off, 32);
                    const int   oe = __shfl_xor(be, off, 32);
                    if (ov > bv || (ov == bv && oi < bi)) { bv = ov; bi = oi; be = oe; }
                }
                if (lane == 0) {
                    outV[iter] = bv; outIp[iter] = bi;
                    sm.p2.V[wave][be] = -1e30f;   // remove winner (in-order LDS within wave)
                }
            }
        }
        __syncthreads();
    }
}

// ---------------- kernel 4: merge slices, softmax, class scatter, normalize ----------------
// one wave per query; grid = NQ/8 blocks of 256
__global__ __launch_bounds__(256)
void classify_kernel(const float* __restrict__ topV, const int* __restrict__ topI,
                     const int* __restrict__ labels, float* __restrict__ out) {
    __shared__ float hist[8][CCLS];
    const int wave = threadIdx.x >> 5, lane = threadIdx.x & 31;
    const int q = blockIdx.x * 8 + wave;

    for (int c = lane; c < CCLS; c += 32) hist[wave][c] = 0.f;

    // 128 candidates (4 slices x 32), 4 per lane
    const float* pv = topV + (size_t)q * (NSLICE * KTOP);
    const int*   pi = topI + (size_t)q * (NSLICE * KTOP);
    float cv[4]; int ci[4];
    #pragma unroll
    for (int j = 0; j < 4; ++j) { cv[j] = pv[lane + 32 * j]; ci[j] = pi[lane + 32 * j]; }

    unsigned used = 0;
    float myV = -1e30f; int myI = 0; float vmax = 0.f;
    for (int iter = 0; iter < KTOP; ++iter) {
        float bv = -1e30f; int bi = 0x7fffffff; int bl = lane, bs = 0;
        #pragma unroll
        for (int j = 0; j < 4; ++j) {
            const bool ok = !(used & (1u << j));
            if (ok && (cv[j] > bv || (cv[j] == bv && ci[j] < bi))) { bv = cv[j]; bi = ci[j]; bs = j; }
        }
        #pragma unroll
        for (int off = 16; off > 0; off >>= 1) {
            const float ov = __shfl_xor(bv, off, 32);
            const int   oi = __shfl_xor(bi, off, 32);
            const int   ol = __shfl_xor(bl, off, 32);
            const int   os = __shfl_xor(bs, off, 32);
            if (ov > bv || (ov == bv && oi < bi)) { bv = ov; bi = oi; bl = ol; bs = os; }
        }
        if (iter == 0) vmax = bv;
        if (lane == bl) used |= (1u << bs);      // winner's owner retires the slot
        if (lane == iter) { myV = bv; myI = bi; } // lane k keeps k-th largest
    }

    // softmax over the 32 winners (K == wave size)
    float w = __expf((myV - vmax) * (1.0f / TAU_F));
    float wsum = w;
    #pragma unroll
    for (int off = 16; off > 0; off >>= 1) wsum += __shfl_xor(wsum, off, 32);
    w /= wsum;

    const int y = labels[myI];
    atomicAdd(&hist[wave][y], w);

    // row-sum (== 1 up to fp error, reference divides anyway) and write
    float s = 0.f;
    for (int c = lane; c < CCLS; c += 32) s += hist[wave][c];
    #pragma unroll
    for (int off = 16; off > 0; off >>= 1) s += __shfl_xor(s, off, 32);
    const float inv = 1.0f / fmaxf(s, 1e-8f);
    float* orow = out + (size_t)q * CCLS;
    for (int c = lane; c < CCLS; c += 32) orow[c] = hist[wave][c] * inv;
}

// ---------------- launcher ----------------
extern "C" void kernel_launch(void* const* d_in, const int* in_sizes, int n_in,
                              void* d_out, int out_size, void* d_ws, size_t ws_size,
                              hipStream_t stream) {
    (void)in_sizes; (void)n_in; (void)out_size; (void)ws_size;
    const float* q      = (const float*)d_in[0];
    const float* feats  = (const float*)d_in[1];
    const int*   labels = (const int*)d_in[2];
    float* out = (float*)d_out;

    char* ws = (char*)d_ws;
    unsigned short* qn = (unsigned short*)ws;                                  // 2 MB
    unsigned short* fnrm = (unsigned short*)(ws + (size_t)NQ * DIM * 2);       // 102.5 MB
    size_t off = (size_t)NQ * DIM * 2 + (size_t)NFPAD * DIM * 2;
    float* topV = (float*)(ws + off);                                          // 1 MB
    int*   topI = (int*)(ws + off + (size_t)NQ * NSLICE * KTOP * 4);           // 1 MB

    nrm_bf16_kernel<<<NQ / 8, 256, 0, stream>>>(q, qn, NQ, NQ);
    nrm_bf16_kernel<<<NFPAD / 8, 256, 0, stream>>>(feats, fnrm, NF, NFPAD);

    dim3 grid(NQ / QTILE, NSLICE);
    sim_topk_kernel<<<grid, 256, 0, stream>>>(qn, fnrm, topV, topI);

    classify_kernel<<<NQ / 8, 256, 0, stream>>>(topV, topI, labels, out);
}
